// Network_22668837388852
// MI455X (gfx1250) — matrix operand order repaired and checked by reading, not compiled, not run on hardware
//
#include <hip/hip_runtime.h>
#include <hip/hip_bf16.h>
#include <math.h>

typedef __attribute__((ext_vector_type(8)))  _Float16 v8h;
typedef __attribute__((ext_vector_type(16))) _Float16 v16h;
typedef __attribute__((ext_vector_type(8)))  float    v8f;

#define N_RAYS 1024
#define NS     128          // samples per ray (coarse and fine)
#define SA     328          // bufA stride in halfs (>=320, odd*8)
#define SB     392          // bufB stride in halfs (>=384, odd*8)
#define SE     72           // encX stride in halfs (>=64, odd*8)

// ---- LDS layout (bytes) ----
#define L_BUFA 0                       // 128*328*2 = 83968
#define L_BUFB 83968                   // 128*392*2 = 100352
#define L_ENCX 184320                  // 128*72*2  = 18432
#define L_XYZ  202752                  // 128*4*4   = 2048 (xyz padded to 4)
#define L_ZC   204800                  // 132*4
#define L_ZF   205328
#define L_CDF  205856
#define L_SIG  206384                  // 128*4
#define L_W    206896                  // 128*4
#define L_RAYC 207408                  // 16*4
#define L_OUT3 207472                  // 4*4
#define SMEM_BYTES 207488

// ---- workspace layout (offsets in halfs) for transposed f16 weights ----
#define OFF_PW0   0u        // 64x128
#define OFF_PW1   8192u     // 128x128
#define OFF_PW2   24576u    // 128x128
#define OFF_FW0   40960u    // 64x256
#define OFF_FWM0  57344u    // 256x256
#define OFF_FWM1  122880u
#define OFF_FWM2  188416u
#define OFF_FWS   253952u   // 320x256
#define OFF_FWP0  335872u
#define OFF_FWP1  401408u
#define OFF_FWP2  466944u
#define OFF_WFEAT 532480u   // 256x256
#define OFF_WVIEW 598016u   // 384x128  (ends at 647168 halfs = 1.23 MB)

struct Params {
    const float* rays;
    const float* emb;
    const float* pb0; const float* pb1; const float* pb2;
    const float* pWo; const float* pbo;
    const float* fb0; const float* fbm; const float* fbs; const float* fbp;
    const float* Wsig; const float* bsig; const float* bfeat; const float* bview;
    const float* Wrgb; const float* brgb;
    const float* minp; const float* maxp;
    const _Float16 *pW0t, *pW1t, *pW2t, *fW0t;
    const _Float16 *fWm0t, *fWm1t, *fWm2t, *fWst;
    const _Float16 *fWp0t, *fWp1t, *fWp2t, *Wfeatt, *Wviewt;
    float* out;
};

// Transpose-convert f32 W[K][N] (row-major) -> f16 Wt[N][Kpad], zero-padded K.
__global__ void convW_kernel(const float* __restrict__ W, _Float16* __restrict__ out,
                             int K, int N, int Kpad) {
    int i = blockIdx.x * blockDim.x + threadIdx.x;
    if (i >= N * Kpad) return;
    int n = i / Kpad, k = i - n * Kpad;
    out[i] = (k < K) ? (_Float16)W[(size_t)k * N + n] : (_Float16)0.0f;
}

// ---- 128 x (NT*16) = 128xK @ Kx(NT*16) GEMM on one workgroup (8 wave32) ----
// Each wave owns a 16-row strip and ALL NT column tiles (acc resident in VGPRs).
// First K-step peeled with inline-zero C (no accumulator-init movs); bias is
// added in the epilogue. K-outer loop: A fragment loaded once per K-step
// (2 x ds_load_b128), then NT independent WMMAs over NT B fragments (global,
// L2-hot, N-major/K-contig). A/B/C VGPR layouts follow the CDNA5 ISA tables.
// Requires full 256-thread convergence (EXEC all ones).
template <int NT>
__device__ __forceinline__ void gemm128(const _Float16* __restrict__ Ain, int strideA,
                                        const _Float16* __restrict__ Wt, int Kpad,
                                        const float* __restrict__ bias,
                                        _Float16* __restrict__ Out, int strideO,
                                        int K, int relu) {
    const int lane = threadIdx.x & 31;
    const int m0   = (threadIdx.x >> 5) << 4;   // wave -> 16-row strip
    const int lrow = lane & 15;
    const int hi   = lane >> 4;                 // K-half selector per ISA layout

    const _Float16* arow = Ain + (m0 + lrow) * strideA + (hi << 3);
    const _Float16* wrow = Wt + (size_t)lrow * Kpad + (hi << 3);

    v8f acc[NT];

    // ---- peeled K-step 0: C = inline zero ----
    {
        v8h a0 = *(const v8h*)(arow);
        v8h a1 = *(const v8h*)(arow + 16);
        v16h af = __builtin_shufflevector(a0, a1, 0,1,2,3,4,5,6,7,8,9,10,11,12,13,14,15);
        const v8f zacc = {};
#pragma unroll
        for (int t = 0; t < NT; ++t) {
            const _Float16* bp = wrow + (size_t)(t * 16) * Kpad;
            v8h b0 = *(const v8h*)(bp);
            v8h b1 = *(const v8h*)(bp + 16);
            v16h bf = __builtin_shufflevector(b0, b1, 0,1,2,3,4,5,6,7,8,9,10,11,12,13,14,15);
            acc[t] = __builtin_amdgcn_wmma_f32_16x16x32_f16(false, af, false, bf,
                                                            (short)0, zacc, false, false);
        }
    }

    for (int k0 = 32; k0 < K; k0 += 32) {
        // A fragment: row m0+lrow, K chunks {k0+hi*8 .. +7} and {+16 .. +23}
        v8h a0 = *(const v8h*)(arow + k0);
        v8h a1 = *(const v8h*)(arow + k0 + 16);
        v16h af = __builtin_shufflevector(a0, a1, 0,1,2,3,4,5,6,7,8,9,10,11,12,13,14,15);
#pragma unroll
        for (int t = 0; t < NT; ++t) {
            // B fragment: col t*16+lrow, K chunks {k0+hi*8 .. +7} and {+16 .. +23}
            const _Float16* bp = wrow + (size_t)(t * 16) * Kpad + k0;
            v8h b0 = *(const v8h*)(bp);
            v8h b1 = *(const v8h*)(bp + 16);
            v16h bf = __builtin_shufflevector(b0, b1, 0,1,2,3,4,5,6,7,8,9,10,11,12,13,14,15);
            acc[t] = __builtin_amdgcn_wmma_f32_16x16x32_f16(false, af, false, bf,
                                                            (short)0, acc[t], false, false);
        }
    }

    // ---- epilogue: bias add (+ ReLU) + f16 store per C/D layout ----
#pragma unroll
    for (int t = 0; t < NT; ++t) {
        float b = bias[t * 16 + lrow];
#pragma unroll
        for (int j = 0; j < 8; ++j) {
            float v = acc[t][j] + b;
            if (relu) v = fmaxf(v, 0.0f);
            Out[(m0 + j + (hi << 3)) * strideO + t * 16 + lrow] = (_Float16)v;
        }
    }
}

// posenc element: [x0,x1,x2, per f: sin(x*2^f)*3, cos(x*2^f)*3]
__device__ __forceinline__ float penc_elem(const float* x3, int c) {
    if (c < 3) return x3[c];
    int f = (c - 3) / 6, rr = (c - 3) % 6, comp = rr % 3;
    float ang = x3[comp] * (float)(1 << f);
    return (rr < 3) ? sinf(ang) : cosf(ang);
}

__global__ __launch_bounds__(256) void nerf_render_kernel(Params P) {
    extern __shared__ char smem[];
    _Float16* bufA = (_Float16*)(smem + L_BUFA);
    _Float16* bufB = (_Float16*)(smem + L_BUFB);
    _Float16* encX = (_Float16*)(smem + L_ENCX);
    float* xyz  = (float*)(smem + L_XYZ);   // [128][4]
    float* zc   = (float*)(smem + L_ZC);    // 129 coarse bin edges
    float* zf   = (float*)(smem + L_ZF);    // 129 fine bin edges
    float* cdf  = (float*)(smem + L_CDF);   // 129
    float* sig  = (float*)(smem + L_SIG);   // 128
    float* wsmp = (float*)(smem + L_W);     // 128
    float* rayc = (float*)(smem + L_RAYC);  // per-ray constants
    float* out3 = (float*)(smem + L_OUT3);

    const int tid = threadIdx.x;
    const int ray = blockIdx.x;

    // ---- per-ray constants ----
    if (tid == 0) {
        const float* rp = P.rays + ray * 12;
        float d0 = rp[3], d1 = rp[4], d2 = rp[5];
        float dn = sqrtf(d0 * d0 + d1 * d1 + d2 * d2);
        rayc[0] = rp[0]; rayc[1] = rp[1]; rayc[2] = rp[2];
        rayc[3] = d0;    rayc[4] = d1;    rayc[5] = d2;
        rayc[6] = d0 / dn; rayc[7] = d1 / dn; rayc[8] = d2 / dn;
        rayc[9]  = fmaxf(rp[6], 1e-8f);   // near
        rayc[10] = rp[7];                 // far
        rayc[11] = rp[8];                 // t
        rayc[12] = dn;                    // |d|
        rayc[13] = rp[9];                 // emb id (exact small int)
        out3[0] = 0.f; out3[1] = 0.f; out3[2] = 0.f;
    }

    // Warm L2/L0 for the fine trunk's first weight matrices while the coarse
    // pass runs (global_prefetch_b8, one cacheline per address).
    for (int i = tid; i < 1024; i += 256) {
        __builtin_prefetch(P.fWm0t + (size_t)i * 64, 0, 3);  // 128 KB
        if (i < 256) __builtin_prefetch(P.fW0t + (size_t)i * 64, 0, 3);  // 32 KB
    }
    __syncthreads();

    // ---- coarse z bins + sample midpoints ----
    if (tid <= NS) zc[tid] = rayc[9] + (rayc[10] - rayc[9]) * (tid * (1.0f / NS));
    __syncthreads();
    for (int idx = tid; idx < NS * 3; idx += 256) {
        int p = idx / 3, c = idx - p * 3;
        float m = 0.5f * (zc[p] + zc[p + 1]);
        xyz[p * 4 + c] = rayc[c] + rayc[3 + c] * m;
    }
    __syncthreads();
    for (int idx = tid; idx < NS * 64; idx += 256) {
        int p = idx >> 6, c = idx & 63;
        float v = (c == 63) ? 0.0f : penc_elem(&xyz[p * 4], c);
        encX[p * SE + c] = (_Float16)v;
    }
    __syncthreads();

    // ---- coarse MLP: 64->128->128->128 ----
    gemm128<8>(encX, SE, P.pW0t, 64,  P.pb0, bufA, SA, 64,  1); __syncthreads();
    gemm128<8>(bufA, SA, P.pW1t, 128, P.pb1, bufB, SB, 128, 1); __syncthreads();
    gemm128<8>(bufB, SB, P.pW2t, 128, P.pb2, bufA, SA, 128, 1); __syncthreads();
    if (tid < NS) {  // sigma_c = h @ pWo + pbo
        float s = P.pbo[0];
        for (int k = 0; k < 128; ++k) s += (float)bufA[tid * SA + k] * P.pWo[k];
        sig[tid] = s;
    }
    __syncthreads();

    // ---- raw2weights (coarse) + CDF, serial per ray ----
    if (tid == 0) {
        float T = 1.0f;
        for (int i = 0; i < NS; ++i) {
            float dist = (zc[i + 1] - zc[i]) * rayc[12];
            float a = 1.0f - expf(-fmaxf(sig[i], 0.0f) * dist);
            wsmp[i] = a * T;
            T *= (1.0f - a + 1e-10f);
        }
        float s = 0.0f;
        for (int i = 0; i < NS; ++i) s += wsmp[i] + 1e-5f;
        float c = 0.0f; cdf[0] = 0.0f;
        for (int i = 0; i < NS; ++i) { c += (wsmp[i] + 1e-5f) / s; cdf[i + 1] = c; }
    }
    __syncthreads();

    // ---- inverse-CDF resampling (129 fine bin edges) ----
    if (tid <= NS) {
        float u = tid * (1.0f / NS);
        int lo = 0, hh = NS + 1;                       // searchsorted right on cdf[0..128]
        while (lo < hh) { int m = (lo + hh) >> 1; if (cdf[m] <= u) lo = m + 1; else hh = m; }
        int below = min(max(lo - 1, 0), NS);
        int above = min(lo, NS);
        float c0 = cdf[below], c1 = cdf[above];
        float den = (c1 - c0 < 1e-5f) ? 1.0f : (c1 - c0);
        float t = (u - c0) / den;
        zf[tid] = zc[below] + t * (zc[above] - zc[below]);
    }
    __syncthreads();

    // ---- fine sample positions + encodings ----
    for (int idx = tid; idx < NS * 3; idx += 256) {
        int p = idx / 3, c = idx - p * 3;
        float m = 0.5f * (zf[p] + zf[p + 1]);
        xyz[p * 4 + c] = rayc[c] + rayc[3 + c] * m;
    }
    __syncthreads();
    for (int idx = tid; idx < NS * 64; idx += 256) {
        int p = idx >> 6, c = idx & 63;
        float v = (c == 63) ? 0.0f : penc_elem(&xyz[p * 4], c);
        _Float16 h = (_Float16)v;
        encX[p * SE + c] = h;
        bufA[p * SA + 256 + c] = h;   // skip-connection columns for fWs (K=320)
    }
    // view-head extra inputs -> bufB cols 256..383: [enc_d(27), emb(48), enc_t(13), enc_app(39), pad]
    for (int idx = tid; idx < NS * 128; idx += 256) {
        int p = idx >> 7, c = idx & 127;
        float v = 0.0f;
        if (c < 27) {                                  // posenc(viewdir), F=4
            if (c < 3) v = rayc[6 + c];
            else {
                int f = (c - 3) / 6, rr = (c - 3) % 6, comp = rr % 3;
                float ang = rayc[6 + comp] * (float)(1 << f);
                v = (rr < 3) ? sinf(ang) : cosf(ang);
            }
        } else if (c < 75) {                           // appearance embedding
            v = P.emb[(int)rayc[13] * 48 + (c - 27)];
        } else if (c < 88) {                           // posenc(t), F=6, 1-D
            int cc = c - 75;
            if (cc == 0) v = rayc[11];
            else {
                int f = (cc - 1) >> 1;
                float ang = rayc[11] * (float)(1 << f);
                v = ((cc - 1) & 1) ? cosf(ang) : sinf(ang);
            }
        } else if (c < 127) {                          // posenc((x-min)/(max-min)), F=6
            int ca = c - 88;
            int comp = (ca < 3) ? ca : ((ca - 3) % 6) % 3;
            float xn = (xyz[p * 4 + comp] - P.minp[comp]) / (P.maxp[comp] - P.minp[comp]);
            if (ca < 3) v = xn;
            else {
                int f = (ca - 3) / 6, rr = (ca - 3) % 6;
                float ang = xn * (float)(1 << f);
                v = (rr < 3) ? sinf(ang) : cosf(ang);
            }
        }
        bufB[p * SB + 256 + c] = (_Float16)v;
    }
    __syncthreads();

    // ---- fine MLP trunk ----
    gemm128<16>(encX, SE, P.fW0t,  64,  P.fb0,       bufB, SB, 64,  1); __syncthreads();
    gemm128<16>(bufB, SB, P.fWm0t, 256, P.fbm + 0,   bufA, SA, 256, 1); __syncthreads();
    gemm128<16>(bufA, SA, P.fWm1t, 256, P.fbm + 256, bufB, SB, 256, 1); __syncthreads();
    gemm128<16>(bufB, SB, P.fWm2t, 256, P.fbm + 512, bufA, SA, 256, 1); __syncthreads();
    // skip concat lives in bufA cols 256..319
    gemm128<16>(bufA, SA, P.fWst,  320, P.fbs,       bufB, SB, 320, 1); __syncthreads();
    gemm128<16>(bufB, SB, P.fWp0t, 256, P.fbp + 0,   bufA, SA, 256, 1); __syncthreads();
    gemm128<16>(bufA, SA, P.fWp1t, 256, P.fbp + 256, bufB, SB, 256, 1); __syncthreads();
    gemm128<16>(bufB, SB, P.fWp2t, 256, P.fbp + 512, bufA, SA, 256, 1); __syncthreads();

    if (tid < NS) {  // sigma_f = h @ Wsig + bsig
        float s = P.bsig[0];
        for (int k = 0; k < 256; ++k) s += (float)bufA[tid * SA + k] * P.Wsig[k];
        sig[tid] = s;
    }
    __syncthreads();
    // feat (no relu) -> bufB cols 0..255; view extras already at 256..383
    gemm128<16>(bufA, SA, P.Wfeatt, 256, P.bfeat, bufB, SB, 256, 0); __syncthreads();
    gemm128<8>(bufB, SB, P.Wviewt, 384, P.bview, bufA, SA, 384, 1); __syncthreads();

    // ---- fine raw2weights ----
    if (tid == 0) {
        float T = 1.0f;
        for (int i = 0; i < NS; ++i) {
            float dist = (zf[i + 1] - zf[i]) * rayc[12];
            float a = 1.0f - expf(-fmaxf(sig[i], 0.0f) * dist);
            wsmp[i] = a * T;
            T *= (1.0f - a + 1e-10f);
        }
    }
    __syncthreads();

    // ---- rgb head + weighted integration ----
    if (tid < NS) {
        float a0 = P.brgb[0], a1 = P.brgb[1], a2 = P.brgb[2];
        for (int k = 0; k < 128; ++k) {
            float h = (float)bufA[tid * SA + k];
            a0 += h * P.Wrgb[k * 3 + 0];
            a1 += h * P.Wrgb[k * 3 + 1];
            a2 += h * P.Wrgb[k * 3 + 2];
        }
        float w = wsmp[tid];
        atomicAdd(&out3[0], w * (1.0f / (1.0f + expf(-a0))));
        atomicAdd(&out3[1], w * (1.0f / (1.0f + expf(-a1))));
        atomicAdd(&out3[2], w * (1.0f / (1.0f + expf(-a2))));
    }
    __syncthreads();
    if (tid < 3) P.out[ray * 3 + tid] = out3[tid];
}

extern "C" void kernel_launch(void* const* d_in, const int* in_sizes, int n_in,
                              void* d_out, int out_size, void* d_ws, size_t ws_size,
                              hipStream_t stream) {
    (void)in_sizes; (void)n_in; (void)out_size; (void)ws_size;
    const float* rays  = (const float*)d_in[0];
    const float* emb   = (const float*)d_in[1];
    const float* pW0   = (const float*)d_in[2];
    const float* pb0   = (const float*)d_in[3];
    const float* pW1   = (const float*)d_in[4];
    const float* pb1   = (const float*)d_in[5];
    const float* pW2   = (const float*)d_in[6];
    const float* pb2   = (const float*)d_in[7];
    const float* pWo   = (const float*)d_in[8];
    const float* pbo   = (const float*)d_in[9];
    const float* fW0   = (const float*)d_in[10];
    const float* fb0   = (const float*)d_in[11];
    const float* fWm   = (const float*)d_in[12];
    const float* fbm   = (const float*)d_in[13];
    const float* fWs   = (const float*)d_in[14];
    const float* fbs   = (const float*)d_in[15];
    const float* fWp   = (const float*)d_in[16];
    const float* fbp   = (const float*)d_in[17];
    const float* Wsig  = (const float*)d_in[18];
    const float* bsig  = (const float*)d_in[19];
    const float* Wfeat = (const float*)d_in[20];
    const float* bfeat = (const float*)d_in[21];
    const float* Wview = (const float*)d_in[22];
    const float* bview = (const float*)d_in[23];
    const float* Wrgb  = (const float*)d_in[24];
    const float* brgb  = (const float*)d_in[25];
    const float* minp  = (const float*)d_in[26];
    const float* maxp  = (const float*)d_in[27];

    _Float16* ws = (_Float16*)d_ws;

    struct Conv { const float* src; unsigned off; int K, N, Kpad; };
    const Conv tab[] = {
        { pW0,          OFF_PW0,   63,  128, 64  },
        { pW1,          OFF_PW1,   128, 128, 128 },
        { pW2,          OFF_PW2,   128, 128, 128 },
        { fW0,          OFF_FW0,   63,  256, 64  },
        { fWm + 0,      OFF_FWM0,  256, 256, 256 },
        { fWm + 65536,  OFF_FWM1,  256, 256, 256 },
        { fWm + 131072, OFF_FWM2,  256, 256, 256 },
        { fWs,          OFF_FWS,   319, 256, 320 },
        { fWp + 0,      OFF_FWP0,  256, 256, 256 },
        { fWp + 65536,  OFF_FWP1,  256, 256, 256 },
        { fWp + 131072, OFF_FWP2,  256, 256, 256 },
        { Wfeat,        OFF_WFEAT, 256, 256, 256 },
        { Wview,        OFF_WVIEW, 383, 128, 384 },
    };
    for (const Conv& c : tab) {
        int total = c.N * c.Kpad;
        convW_kernel<<<(total + 255) / 256, 256, 0, stream>>>(c.src, ws + c.off,
                                                              c.K, c.N, c.Kpad);
    }

    Params P;
    P.rays = rays; P.emb = emb;
    P.pb0 = pb0; P.pb1 = pb1; P.pb2 = pb2; P.pWo = pWo; P.pbo = pbo;
    P.fb0 = fb0; P.fbm = fbm; P.fbs = fbs; P.fbp = fbp;
    P.Wsig = Wsig; P.bsig = bsig; P.bfeat = bfeat; P.bview = bview;
    P.Wrgb = Wrgb; P.brgb = brgb; P.minp = minp; P.maxp = maxp;
    P.pW0t = ws + OFF_PW0;  P.pW1t = ws + OFF_PW1;  P.pW2t = ws + OFF_PW2;
    P.fW0t = ws + OFF_FW0;
    P.fWm0t = ws + OFF_FWM0; P.fWm1t = ws + OFF_FWM1; P.fWm2t = ws + OFF_FWM2;
    P.fWst = ws + OFF_FWS;
    P.fWp0t = ws + OFF_FWP0; P.fWp1t = ws + OFF_FWP1; P.fWp2t = ws + OFF_FWP2;
    P.Wfeatt = ws + OFF_WFEAT; P.Wviewt = ws + OFF_WVIEW;
    P.out = (float*)d_out;

    hipFuncSetAttribute(reinterpret_cast<const void*>(nerf_render_kernel),
                        hipFuncAttributeMaxDynamicSharedMemorySize, SMEM_BYTES);
    nerf_render_kernel<<<N_RAYS, 256, SMEM_BYTES, stream>>>(P);
}
